// MixtureOfExperts_82291573391898
// MI455X (gfx1250) — matrix operand order repaired and checked by reading, not compiled, not run on hardware
//
#include <hip/hip_runtime.h>

#define DIM   1024
#define NEXP  8
#define HID   2730
#define HPAD  2816                  // 22 * 128, zero-padded hidden
#define NTOK  8192                  // 4 * 2048
#define OUT_ELEMS (NTOK * DIM)      // 8388608 floats, aux scalar follows
#define TILES_PER_E (NTOK / 16)     // 512
#define XSS 1040                    // LDS x-tile row stride (bf16 elems): 512 + 8 pad + 512 + 8 pad
#define HP  136                     // padded LDS row stride (bf16 elems) for h tile
#define NCHUNK 22                   // HPAD / 128

typedef __attribute__((ext_vector_type(16))) __bf16       v16bf;
typedef __attribute__((ext_vector_type(8)))  float        v8f;
typedef __attribute__((ext_vector_type(4))) unsigned int  u32x4;
typedef __attribute__((ext_vector_type(8)))  int          i32x8;
typedef __attribute__((ext_vector_type(4)))  int          i32x4;

union BF16x16 { v16bf v; u32x4 q[2]; unsigned short s[16]; };

__device__ __forceinline__ unsigned short f2bf(float f) {
  unsigned u = __float_as_uint(f);
  u += 0x7FFFu + ((u >> 16) & 1u);          // round-to-nearest-even
  return (unsigned short)(u >> 16);
}

__device__ __forceinline__ v8f vzero8() {
  v8f z = {0.f, 0.f, 0.f, 0.f, 0.f, 0.f, 0.f, 0.f};
  return z;
}

// ---------------------------------------------------------------- zero/init
__global__ void moe_zero(float* __restrict__ out, int* __restrict__ cnt,
                         float* __restrict__ psum) {
  size_t i = (size_t)blockIdx.x * blockDim.x + threadIdx.x;
  size_t stride = (size_t)gridDim.x * blockDim.x;
  for (; i < (size_t)OUT_ELEMS; i += stride) out[i] = 0.f;
  if (blockIdx.x == 0 && threadIdx.x < NEXP) {
    cnt[threadIdx.x] = 0;
    psum[threadIdx.x] = 0.f;
  }
}

// ------------------------------------------------- weight f32 -> padded bf16
__global__ void moe_cvt_w(const float* __restrict__ w1, const float* __restrict__ w2,
                          const float* __restrict__ w3,
                          unsigned short* __restrict__ w1b,
                          unsigned short* __restrict__ w2b,
                          unsigned short* __restrict__ w3b) {
  const size_t NW = (size_t)NEXP * HPAD * DIM;
  size_t stride = (size_t)gridDim.x * blockDim.x;
  for (size_t idx = (size_t)blockIdx.x * blockDim.x + threadIdx.x; idx < NW; idx += stride) {
    size_t e   = idx / ((size_t)HPAD * DIM);
    size_t rem = idx % ((size_t)HPAD * DIM);
    // w1/w3: [e][h(HPAD)][d]
    size_t h = rem / DIM, d = rem % DIM;
    unsigned short v1 = 0, v3 = 0;
    if (h < HID) {
      size_t s = (e * HID + h) * DIM + d;
      v1 = f2bf(w1[s]); v3 = f2bf(w3[s]);
    }
    w1b[idx] = v1; w3b[idx] = v3;
    // w2: [e][d2][h2(HPAD)]
    size_t d2 = rem / HPAD, h2 = rem % HPAD;
    unsigned short v2 = 0;
    if (h2 < HID) v2 = f2bf(w2[(e * DIM + d2) * HID + h2]);
    w2b[idx] = v2;
  }
}

// ---------------------------------------------------------------- router
__global__ void __launch_bounds__(256) moe_router(
    const float* __restrict__ x, const float* __restrict__ gw,
    int* __restrict__ cnt, float* __restrict__ psum,
    int* __restrict__ tlist, float* __restrict__ wlist) {
  __shared__ float ps[NEXP];
  const int tid = threadIdx.x, lane = tid & 31, wave = tid >> 5;
  if (tid < NEXP) ps[tid] = 0.f;
  __syncthreads();

  const int t = blockIdx.x * 8 + wave;       // one wave32 per token
  float a[NEXP];
#pragma unroll
  for (int e = 0; e < NEXP; ++e) a[e] = 0.f;
  const float* xr = x + (size_t)t * DIM;
  for (int d = lane; d < DIM; d += 32) {
    float xv = xr[d];
#pragma unroll
    for (int e = 0; e < NEXP; ++e) a[e] += xv * gw[e * DIM + d];
  }
#pragma unroll
  for (int e = 0; e < NEXP; ++e)
    for (int off = 16; off > 0; off >>= 1) a[e] += __shfl_xor(a[e], off, 32);

  if (lane == 0) {
    float mx = a[0];
#pragma unroll
    for (int e = 1; e < NEXP; ++e) mx = fmaxf(mx, a[e]);
    float p[NEXP]; float s = 0.f;
#pragma unroll
    for (int e = 0; e < NEXP; ++e) { p[e] = __expf(a[e] - mx); s += p[e]; }
    float inv = 1.f / s;
#pragma unroll
    for (int e = 0; e < NEXP; ++e) {
      p[e] *= inv;
      atomicAdd(&ps[e], p[e]);
    }
    int i0 = 0;
#pragma unroll
    for (int e = 1; e < NEXP; ++e) if (p[e] > p[i0]) i0 = e;
    int i1 = (i0 == 0) ? 1 : 0;
#pragma unroll
    for (int e = 0; e < NEXP; ++e) if (e != i0 && p[e] > p[i1]) i1 = e;
    float sw = p[i0] + p[i1];
    float w0 = p[i0] / sw, w1 = p[i1] / sw;
    int pos0 = atomicAdd(&cnt[i0], 1);
    tlist[i0 * NTOK + pos0] = t; wlist[i0 * NTOK + pos0] = w0;
    int pos1 = atomicAdd(&cnt[i1], 1);
    tlist[i1 * NTOK + pos1] = t; wlist[i1 * NTOK + pos1] = w1;
  }
  __syncthreads();
  if (tid < NEXP) atomicAdd(&psum[tid], ps[tid]);
}

// --------------------------------------- gather routed tokens as bf16 rows
__global__ void __launch_bounds__(256) moe_gather_x(
    const float* __restrict__ x, const int* __restrict__ cnt,
    const int* __restrict__ tlist, unsigned short* __restrict__ xg) {
  int row = blockIdx.x * 8 + (threadIdx.x >> 5);   // slot across all experts
  int lane = threadIdx.x & 31;
  int e = row >> 13;                                // / NTOK
  int i = row & (NTOK - 1);
  int c = cnt[e];
  int cpad = (c + 15) & ~15;                        // zero-fill to tile boundary
  if (i >= cpad) return;
  unsigned short* dst = xg + (size_t)row * DIM;
  if (i < c) {
    const float* src = x + (size_t)tlist[e * NTOK + i] * DIM;
    for (int d = lane; d < DIM; d += 32) dst[d] = f2bf(src[d]);
  } else {
    for (int d = lane; d < DIM; d += 32) dst[d] = 0;
  }
}

// ---------------------------------------------------------------- expert MLP
__global__ void __launch_bounds__(256) moe_expert(
    const unsigned short* __restrict__ xg,  const unsigned short* __restrict__ w1b,
    const unsigned short* __restrict__ w2b, const unsigned short* __restrict__ w3b,
    const int* __restrict__ cnt,  const int* __restrict__ tlist,
    const float* __restrict__ wlist, float* __restrict__ out) {
  __shared__ __align__(16) unsigned short xs[16 * XSS];  // TDM-loaded x tile (bf16)
  __shared__ __align__(16) unsigned short hs[16 * HP];   // h chunk (bf16)
  __shared__ int   tk_id[16];
  __shared__ float tk_w[16];

  const int e = blockIdx.x / TILES_PER_E;
  const int tile = blockIdx.x % TILES_PER_E;
  const int count = cnt[e];
  const int tbase = tile * 16;
  if (tbase >= count) return;                             // uniform exit

  const int tid = threadIdx.x, lane = tid & 31, wave = tid >> 5;
  const int khalf = lane >> 4;
  const int lmod = lane & 15;

  if (tid < 16) {
    int idx = tbase + tid;
    if (idx < count) { tk_id[tid] = tlist[e * NTOK + idx]; tk_w[tid] = wlist[e * NTOK + idx]; }
    else             { tk_id[tid] = -1;                    tk_w[tid] = 0.f; }
  }

  // --- Tensor Data Mover: pull 16x1024 bf16 tile into LDS with row padding ---
  if (wave == 0) {
    unsigned lds_off = (unsigned)(unsigned long long)(void*)xs;
    unsigned long long ga =
        (unsigned long long)(const void*)(xg + (size_t)(e * NTOK + tbase) * DIM);
    u32x4 g0 = { 1u,                                   // count=1, user mode, no gather
                 lds_off,                              // lds_addr (bytes)
                 (unsigned)ga,                         // global_addr[31:0]
                 (unsigned)((ga >> 32) & 0x1FFFFFFull) | (2u << 30) };  // addr[56:32] | type=2
    i32x8 g1 = { (int)((1u << 16)                      // data_size = 2 bytes
                     | (1u << 20)                      // pad_enable
                     | (7u << 22)                      // pad_interval = 256 DWORDs
                     | (3u << 25)),                    // pad_amount  = 4 DWORDs (8 bf16)
                 (int)(1024u << 16),                   // tensor_dim0[15:0] = 1024
                 (int)(8192u << 16),                   // tensor_dim1[15:0] = 8192
                 (int)(1024u << 16),                   // tile_dim0 = 1024
                 16,                                   // tile_dim1 = 16 rows
                 1024,                                 // tensor_dim0_stride = 1024 elems
                 0, 0 };
    i32x4 gz4 = {0, 0, 0, 0};
    i32x8 gz8 = {0, 0, 0, 0, 0, 0, 0, 0};
    __builtin_amdgcn_tensor_load_to_lds(g0, g1, gz4, gz4, gz8, 0);
    __builtin_amdgcn_s_wait_tensorcnt(0);
  }
  __syncthreads();

  v8f acc2[8];                                            // 16x128 output slice
#pragma unroll
  for (int i = 0; i < 8; ++i) acc2[i] = vzero8();

  for (int c = 0; c < NCHUNK; ++c) {
    const int hb = c * 128;
    // ---- GEMM1: wave's 16-wide hidden tile, K = 1024 (branch-free) ----
    const int hrow = hb + wave * 16 + lmod;               // < HPAD always
    const unsigned short* w1p = w1b + ((size_t)e * HPAD + hrow) * DIM + khalf * 16;
    const unsigned short* w3p = w3b + ((size_t)e * HPAD + hrow) * DIM + khalf * 16;
    v8f a1 = vzero8(), a3 = vzero8();
    for (int k0 = 0; k0 < DIM; k0 += 32) {
      BF16x16 A, B1, B3;
      const int e1 = k0 + khalf * 16;
      const int e2 = e1 + 8;
      A.q[0] = *(const u32x4*)(xs + lmod * XSS + e1 + ((e1 >= 512) ? 8 : 0));
      A.q[1] = *(const u32x4*)(xs + lmod * XSS + e2 + ((e2 >= 512) ? 8 : 0));
      B1.q[0] = *(const u32x4*)(w1p + k0);
      B1.q[1] = *(const u32x4*)(w1p + k0 + 8);
      B3.q[0] = *(const u32x4*)(w3p + k0);
      B3.q[1] = *(const u32x4*)(w3p + k0 + 8);
      __builtin_prefetch(w1p + k0 + 256, 0, 1);
      __builtin_prefetch(w3p + k0 + 256, 0, 1);
      a1 = __builtin_amdgcn_wmma_f32_16x16x32_bf16(false, A.v, false, B1.v,
                                                   (short)0, a1, false, false);
      a3 = __builtin_amdgcn_wmma_f32_16x16x32_bf16(false, A.v, false, B3.v,
                                                   (short)0, a3, false, false);
    }
    // ---- SwiGLU + stage h chunk to LDS ----
    __syncthreads();                                      // prior GEMM2 reads done
    {
      const int m0 = khalf * 8;
      const int hcol = wave * 16 + lmod;
#pragma unroll
      for (int j = 0; j < 8; ++j) {
        float g = a1[j];
        float h = (g / (1.f + __expf(-g))) * a3[j];
        hs[(m0 + j) * HP + hcol] = f2bf(h);
      }
    }
    __syncthreads();
    // ---- GEMM2: y[16 x 128 slice] += h_chunk @ w2^T (branch-free) ----
    const int dbase = wave * 128;
#pragma unroll
    for (int kk = 0; kk < 128; kk += 32) {
      BF16x16 A2;
      const unsigned short* hbp = hs + lmod * HP + kk + khalf * 16;
      A2.q[0] = *(const u32x4*)(hbp);
      A2.q[1] = *(const u32x4*)(hbp + 8);
      const int h0 = hb + kk + khalf * 16;
#pragma unroll
      for (int nt = 0; nt < 8; ++nt) {
        const int d = dbase + nt * 16 + lmod;
        BF16x16 B2;
        const unsigned short* wp = w2b + ((size_t)e * DIM + d) * HPAD + h0;
        B2.q[0] = *(const u32x4*)(wp);
        B2.q[1] = *(const u32x4*)(wp + 8);
        acc2[nt] = __builtin_amdgcn_wmma_f32_16x16x32_bf16(false, A2.v, false, B2.v,
                                                           (short)0, acc2[nt], false, false);
      }
    }
  }

  // ---- scale by routing weight and accumulate into out ----
  {
    const int m0 = khalf * 8;
    const int dcol = wave * 128 + lmod;
#pragma unroll
    for (int j = 0; j < 8; ++j) {
      int m = m0 + j;
      int tok = tk_id[m];
      float wgt = tk_w[m];
      if (tok >= 0 && wgt != 0.f) {
#pragma unroll
        for (int nt = 0; nt < 8; ++nt)
          atomicAdd(&out[(size_t)tok * DIM + dcol + nt * 16], acc2[nt][j] * wgt);
      }
    }
  }
}

// ---------------------------------------------------------------- aux loss
__global__ void moe_aux(const float* __restrict__ psum, float* __restrict__ aux) {
  if (threadIdx.x == 0) {
    float a = 0.f;
#pragma unroll
    for (int e = 0; e < NEXP; ++e) {
      float tpe = psum[e] / (float)NTOK;
      float d = tpe - (1.f / NEXP);
      a += d * d;
    }
    aux[0] = 0.01f * (a / (float)NEXP);
  }
}

// ---------------------------------------------------------------- launcher
extern "C" void kernel_launch(void* const* d_in, const int* in_sizes, int n_in,
                              void* d_out, int out_size, void* d_ws, size_t ws_size,
                              hipStream_t stream) {
  const float* x  = (const float*)d_in[0];
  const float* gw = (const float*)d_in[1];
  const float* w1 = (const float*)d_in[2];
  const float* w2 = (const float*)d_in[3];
  const float* w3 = (const float*)d_in[4];
  float* out = (float*)d_out;

  // workspace layout (requires ~274 MB)
  char* ws = (char*)d_ws;
  const size_t WPAD_BYTES = (size_t)NEXP * HPAD * DIM * 2;   // 46,137,344 per tensor
  int*   cnt   = (int*)(ws);
  float* psum  = (float*)(ws + 32);
  int*   tlist = (int*)(ws + 64);
  float* wlist = (float*)(ws + 64 + (size_t)NEXP * NTOK * sizeof(int));
  unsigned short* w1b = (unsigned short*)(ws + (1u << 20));
  unsigned short* w3b = (unsigned short*)(ws + (1u << 20) + WPAD_BYTES);
  unsigned short* w2b = (unsigned short*)(ws + (1u << 20) + 2 * WPAD_BYTES);
  unsigned short* xg  = (unsigned short*)(ws + (1u << 20) + 3 * WPAD_BYTES);

  moe_zero<<<4096, 256, 0, stream>>>(out, cnt, psum);
  moe_cvt_w<<<4096, 256, 0, stream>>>(w1, w2, w3, w1b, w2b, w3b);
  moe_router<<<NTOK / 8, 256, 0, stream>>>(x, gw, cnt, psum, tlist, wlist);
  moe_gather_x<<<NEXP * NTOK / 8, 256, 0, stream>>>(x, cnt, tlist, xg);
  moe_expert<<<NEXP * TILES_PER_E, 256, 0, stream>>>(xg, w1b, w2b, w3b, cnt, tlist, wlist, out);
  moe_aux<<<1, 32, 0, stream>>>(psum, out + (size_t)OUT_ELEMS);
}
